// SO2EquivariantGraphAttention_33354716021255
// MI455X (gfx1250) — matrix operand order repaired and checked by reading, not compiled, not run on hardware
//
#include <hip/hip_runtime.h>

// ---------------------------------------------------------------------------
// SO2 equivariant graph attention, restructured for MI455X (gfx1250, wave32).
// All big GEMMs run on v_wmma_f32_16x16x32_bf16 (bf16 in, f32 accumulate).
// Edge kernels: M=32 tiles (two A-tiles share each B fragment) + depth-2
// B-fragment software pipeline.
// ---------------------------------------------------------------------------

#define N_NODES   10000
#define N_EDGES   160000
#define NODE_DIM  480
#define SMUL      128     // SCALAR_MUL
#define HEADS     8
#define ALPHA_C   32

typedef __attribute__((ext_vector_type(16))) __bf16 v16bf;
typedef __attribute__((ext_vector_type(8)))  float  v8f;

union BFrag { v16bf v; unsigned u[8]; };

__device__ __forceinline__ unsigned pack_bf2(float lo, float hi) {
  unsigned a = __float_as_uint(lo);
  unsigned b = __float_as_uint(hi);
  a += 0x7FFFu + ((a >> 16) & 1u);   // round-to-nearest-even to bf16
  b += 0x7FFFu + ((b >> 16) & 1u);
  return (a >> 16) | (b & 0xFFFF0000u);
}

// monotone encoding of float for uint atomicMax (handles negatives / -inf)
__device__ __forceinline__ unsigned enc_f(float f) {
  unsigned u = __float_as_uint(f);
  return (u & 0x80000000u) ? ~u : (u | 0x80000000u);
}
__device__ __forceinline__ float dec_f(unsigned u) {
  return __uint_as_float((u & 0x80000000u) ? (u & 0x7FFFFFFFu) : ~u);
}

// hardware-path atomics (global_atomic_add_f32 / global_atomic_max_u32)
__device__ __forceinline__ void atomic_add_f32(float* p, float v) {
  __hip_atomic_fetch_add(p, v, __ATOMIC_RELAXED, __HIP_MEMORY_SCOPE_AGENT);
}
__device__ __forceinline__ void atomic_max_u32(unsigned* p, unsigned v) {
  __hip_atomic_fetch_max(p, v, __ATOMIC_RELAXED, __HIP_MEMORY_SCOPE_AGENT);
}

__device__ __forceinline__ v8f wmma_bf16(v16bf a, v16bf b, v8f c) {
  return __builtin_amdgcn_wmma_f32_16x16x32_bf16(false, a, false, b, (short)0, c,
                                                 false, false);
}

// A-fragment (16x32 bf16) from an LDS tile stored as packed bf16 pairs.
// row_base points at the 64-uint pair row of (tile_row 0) for this fragment.
// ISA layout: lane = half*16 + row; vgpr v<4 -> K = c*32 + half*8 + 2v(+1),
//             vgpr v>=4 -> K = c*32 + 16 + half*8 + 2(v-4)(+1).
__device__ __forceinline__ v16bf lds_a_frag(const unsigned* s_scal, int row0,
                                            int lane, int c) {
  const unsigned* row = s_scal + (row0 + (lane & 15)) * 64;
  int h = lane >> 4;
  BFrag f;
#pragma unroll
  for (int v = 0; v < 4; ++v) f.u[v] = row[c * 16 + h * 4 + v];
#pragma unroll
  for (int v = 0; v < 4; ++v) f.u[4 + v] = row[c * 16 + 8 + h * 4 + v];
  return f.v;
}

// A-fragment from an f32 row-major matrix (convert to bf16 on the fly)
__device__ __forceinline__ v16bf load_a_frag_f32(const float* mat, int ld,
                                                 int row0, int k0, int lane) {
  const float* p0 = mat + (size_t)(row0 + (lane & 15)) * ld + k0 + (lane >> 4) * 8;
  const float* p1 = p0 + 16;
  BFrag f;
#pragma unroll
  for (int v = 0; v < 4; ++v) f.u[v] = pack_bf2(p0[2 * v], p0[2 * v + 1]);
#pragma unroll
  for (int v = 0; v < 4; ++v) f.u[4 + v] = pack_bf2(p1[2 * v], p1[2 * v + 1]);
  return f.v;
}

// B-fragment (32x16 bf16) from packed-bf16 KxN matrix; lane L holds column
// n0+(L&15), K in the same table order as the A fragment.
__device__ __forceinline__ v16bf load_b_frag(const unsigned* mat, int ld_pairs,
                                             int k0, int n0, int lane) {
  const unsigned short* p = reinterpret_cast<const unsigned short*>(mat) + n0 + (lane & 15);
  size_t ld = (size_t)ld_pairs * 2;
  int h = lane >> 4;
  BFrag f;
#pragma unroll
  for (int i = 0; i < 8; ++i) {
    int k = k0 + ((i < 4) ? (h * 8 + 2 * i) : (16 + h * 8 + 2 * (i - 4)));
    f.u[i] = (unsigned)p[(size_t)k * ld] | ((unsigned)p[(size_t)(k + 1) * ld] << 16);
  }
  return f.v;
}

// Gather scal = P_src[src] + P_tgt[tgt] for a 32-edge tile -> LDS bf16 pairs.
// s_idx: [32 src][32 tgt]; s_scal: 32 rows x 64 uint pairs.
__device__ __forceinline__ void gather_scal32(const int* eidx, int e0, int lane,
                                              const float* Psrc, const float* Ptgt,
                                              int* s_idx, unsigned* s_scal) {
  s_idx[lane]      = eidx[e0 + lane];            // src ids
  s_idx[32 + lane] = eidx[N_EDGES + e0 + lane];  // tgt ids
  __syncthreads();
#pragma unroll 4
  for (int i = 0; i < 32; ++i) {
    int qd = i * 32 + lane;    // 1024 float4 quads (32 edges * 32 quads)
    int e  = qd >> 5;
    int j  = qd & 31;
    int s  = s_idx[e];
    int t  = s_idx[32 + e];
    float4 a = *reinterpret_cast<const float4*>(Psrc + (size_t)s * SMUL + j * 4);
    float4 b = *reinterpret_cast<const float4*>(Ptgt + (size_t)t * SMUL + j * 4);
    uint2 pk;
    pk.x = pack_bf2(a.x + b.x, a.y + b.y);
    pk.y = pack_bf2(a.z + b.z, a.w + b.w);
    *reinterpret_cast<uint2*>(&s_scal[e * 64 + j * 2]) = pk;
  }
  __syncthreads();
}

// ---------------------------------------------------------------------------
// Kernels
// ---------------------------------------------------------------------------

__global__ void k_init(float* node_msg, float* den, unsigned* nodemax, float* sig) {
  int tid = blockIdx.x * blockDim.x + threadIdx.x;
  if (tid < N_NODES * SMUL) node_msg[tid] = 0.0f;
  if (tid < N_NODES * HEADS) { den[tid] = 0.0f; nodemax[tid] = 0x007FFFFFu; } // enc(-inf)
  if (tid < 2) sig[tid] = 0.0f;
}

__global__ void k_pack_bf(const float* __restrict__ in, unsigned* __restrict__ out,
                          int npairs) {
  int tid = blockIdx.x * blockDim.x + threadIdx.x;
  if (tid < npairs) out[tid] = pack_bf2(in[2 * tid], in[2 * tid + 1]);
}

// A_comb = W_conv1(128x480) @ alpha_w(480x256); V_comb = W_conv1 @ W_conv2(480x128)
__global__ void k_combine(const float* __restrict__ Wc1, const float* __restrict__ aw,
                          const float* __restrict__ Wc2,
                          unsigned* __restrict__ AcombBf, unsigned* __restrict__ VcombBf) {
  int tid = blockIdx.x * blockDim.x + threadIdx.x;
  if (tid < 128 * 128) {                  // A_comb pairs
    int r = tid >> 7, n = (tid & 127) * 2;
    float d0 = 0.f, d1 = 0.f;
    for (int k = 0; k < 480; ++k) {
      float w = Wc1[r * 480 + k];
      d0 += w * aw[k * 256 + n];
      d1 += w * aw[k * 256 + n + 1];
    }
    AcombBf[tid] = pack_bf2(d0, d1);
  } else if (tid < 128 * 128 + 128 * 64) { // V_comb pairs
    int q = tid - 128 * 128;
    int r = q >> 6, n = (q & 63) * 2;
    float d0 = 0.f, d1 = 0.f;
    for (int k = 0; k < 480; ++k) {
      float w = Wc1[r * 480 + k];
      d0 += w * Wc2[k * 128 + n];
      d1 += w * Wc2[k * 128 + n + 1];
    }
    VcombBf[q] = pack_bf2(d0, d1);
  }
}

// P_src = nf @ W_msg[0:480], P_tgt = nf @ W_msg[480:960]   (WMMA, pipelined)
__global__ __launch_bounds__(32) void k_nodeproj(const float* __restrict__ nf,
                                                 const unsigned* __restrict__ WmsgBf,
                                                 float* __restrict__ Psrc,
                                                 float* __restrict__ Ptgt) {
  int lane = threadIdx.x;
  int row0 = blockIdx.x * 16;
  v8f acc[16];
#pragma unroll
  for (int nt = 0; nt < 16; ++nt) acc[nt] = {};
  v16bf a_cur = load_a_frag_f32(nf, NODE_DIM, row0, 0, lane);
  v16bf b_cur = load_b_frag(WmsgBf, 64, 0, 0, lane);
  for (int c = 0; c < 15; ++c) {          // K = 480
    int cn = (c + 1 < 15) ? (c + 1) : 0;
    v16bf a_next = load_a_frag_f32(nf, NODE_DIM, row0, cn * 32, lane);
#pragma unroll
    for (int nt = 0; nt < 16; ++nt) {     // 8 tiles src + 8 tiles tgt
      int j   = c * 16 + nt + 1;          // next (c,nt) step, wrapped
      int ntn = j & 15;
      int ccn = j >> 4;  if (ccn >= 15) ccn = 0;
      v16bf b_next = load_b_frag(WmsgBf, 64,
                                 ((ntn >= 8) ? 480 : 0) + ccn * 32,
                                 (ntn & 7) * 16, lane);
      acc[nt] = wmma_bf16(a_cur, b_cur, acc[nt]);
      b_cur = b_next;
    }
    a_cur = a_next;
  }
  int crow = (lane >> 4) * 8, ccol = lane & 15;
#pragma unroll
  for (int nt = 0; nt < 16; ++nt) {
    float* dst = (nt >= 8) ? Ptgt : Psrc;
    int n0 = (nt & 7) * 16;
#pragma unroll
    for (int v = 0; v < 8; ++v)
      dst[(size_t)(row0 + crow + v) * SMUL + n0 + ccol] = acc[nt][v];
  }
}

// w1/w2 scalar gates: sum of sigmoid(ef @ ep_w + ep_b) over all edges.
// One wave per 32 edges; a full wave reads one 64-float row coalesced.
__global__ void k_edgescalars(const float* __restrict__ ef,
                              const float* __restrict__ ep1w, const float* __restrict__ ep1b,
                              const float* __restrict__ ep2w, const float* __restrict__ ep2b,
                              float* sig) {
  int lane = threadIdx.x & 31;
  int wave = (blockIdx.x * blockDim.x + threadIdx.x) >> 5;
  int e0 = wave * 32;
  float w1a = ep1w[2 * lane], w1b = ep1w[2 * lane + 1];
  float w2a = ep2w[2 * lane], w2b = ep2w[2 * lane + 1];
  float b1 = ep1b[0], b2 = ep2b[0];
  float s1 = 0.f, s2 = 0.f;
  for (int e = e0; e < e0 + 32; ++e) {
    float2 v = *reinterpret_cast<const float2*>(ef + (size_t)e * 64 + 2 * lane);
    float d1 = v.x * w1a + v.y * w1b;
    float d2 = v.x * w2a + v.y * w2b;
#pragma unroll
    for (int off = 16; off > 0; off >>= 1) {
      d1 += __shfl_xor(d1, off, 32);
      d2 += __shfl_xor(d2, off, 32);
    }
    s1 += 1.0f / (1.0f + __expf(-(d1 + b1)));
    s2 += 1.0f / (1.0f + __expf(-(d2 + b2)));
  }
  if (lane == 0) {
    atomic_add_f32(&sig[0], s1);
    atomic_add_f32(&sig[1], s2);
  }
}

// Pass A: q = scal @ A_comb (WMMA, M=32), LayerNorm + leaky + logits, segment max
__global__ __launch_bounds__(32) void k_edge_logits(
    const int* __restrict__ eidx, const float* __restrict__ Psrc,
    const float* __restrict__ Ptgt, const unsigned* __restrict__ AcombBf,
    const float* __restrict__ alpha_b, const float* __restrict__ ln_g,
    const float* __restrict__ ln_b, const float* __restrict__ alpha_dot,
    const float* __restrict__ sig, float* __restrict__ logits,
    unsigned* __restrict__ nodemax) {
  __shared__ int      s_idx[64];
  __shared__ __align__(16) unsigned s_scal[32 * 64];
  __shared__ float    s_q[32 * 256];
  int lane = threadIdx.x;
  int e0 = blockIdx.x * 32;
  gather_scal32(eidx, e0, lane, Psrc, Ptgt, s_idx, s_scal);

  v16bf aF0[4], aF1[4];
#pragma unroll
  for (int c = 0; c < 4; ++c) {
    aF0[c] = lds_a_frag(s_scal, 0, lane, c);
    aF1[c] = lds_a_frag(s_scal, 16, lane, c);
  }
  int crow = (lane >> 4) * 8, ccol = lane & 15;
  // depth-2 B pipeline over 64 (nt,c) steps
  v16bf bp0 = load_b_frag(AcombBf, 128, 0, 0, lane);       // step 0: c=0,nt=0
  v16bf bp1 = load_b_frag(AcombBf, 128, 32, 0, lane);      // step 1: c=1,nt=0
  for (int nt = 0; nt < 16; ++nt) {       // N = 256
    v8f acc0 = {}, acc1 = {};
#pragma unroll
    for (int c = 0; c < 4; ++c) {         // K = 128
      int jn = (nt * 4 + c + 2) & 63;     // step 2 ahead, wrapped
      v16bf bn = load_b_frag(AcombBf, 128, (jn & 3) * 32, (jn >> 2) * 16, lane);
      acc0 = wmma_bf16(aF0[c], bp0, acc0);
      acc1 = wmma_bf16(aF1[c], bp0, acc1);
      bp0 = bp1;
      bp1 = bn;
    }
#pragma unroll
    for (int v = 0; v < 8; ++v) {
      s_q[(crow + v) * 256 + nt * 16 + ccol]      = acc0[v];
      s_q[(16 + crow + v) * 256 + nt * 16 + ccol] = acc1[v];
    }
  }
  __syncthreads();

  float w1 = sig[0] * (1.0f / N_EDGES);
  for (int r = 0; r < 8; ++r) {           // 256 (edge,head) groups / 32 lanes
    int g = r * 32 + lane;
    int e = g >> 3, h = g & 7;
    const float* qrow = s_q + e * 256 + h * ALPHA_C;
    const float* ab   = alpha_b + h * ALPHA_C;
    float mu = 0.f, m2 = 0.f;
#pragma unroll 8
    for (int i = 0; i < ALPHA_C; ++i) {
      float a = w1 * qrow[i] + ab[i];
      mu += a; m2 += a * a;
    }
    mu *= (1.0f / ALPHA_C);
    float var = fmaxf(m2 * (1.0f / ALPHA_C) - mu * mu, 0.0f);
    float inv = rsqrtf(var + 1e-5f);
    const float* adot = alpha_dot + h * ALPHA_C;
    float logit = 0.f;
#pragma unroll 8
    for (int i = 0; i < ALPHA_C; ++i) {
      float a  = w1 * qrow[i] + ab[i];
      float an = (a - mu) * inv * ln_g[i] + ln_b[i];
      an = (an >= 0.f) ? an : 0.1f * an;
      logit += an * adot[i];
    }
    logits[(size_t)(e0 + e) * HEADS + h] = logit;
    atomic_max_u32(&nodemax[(size_t)s_idx[32 + e] * HEADS + h], enc_f(logit));
  }
}

// Pass B: ex = exp(logit - max), segment-sum denominator (ex overwrites logits)
__global__ void k_den(const int* __restrict__ eidx, const unsigned* __restrict__ nodemax,
                      float* __restrict__ logits, float* __restrict__ den) {
  int tid = blockIdx.x * blockDim.x + threadIdx.x;
  if (tid >= N_EDGES * HEADS) return;
  int e = tid >> 3, h = tid & 7;
  int t = eidx[N_EDGES + e];
  float m = dec_f(nodemax[(size_t)t * HEADS + h]);
  float ex = __expf(logits[tid] - m);
  logits[tid] = ex;
  atomic_add_f32(&den[(size_t)t * HEADS + h], ex);
}

// Pass C: v = scal @ V_comb (WMMA, M=32), scale by w1*w2*alpha, scatter-add
__global__ __launch_bounds__(32) void k_edge_accum(
    const int* __restrict__ eidx, const float* __restrict__ Psrc,
    const float* __restrict__ Ptgt, const unsigned* __restrict__ VcombBf,
    const float* __restrict__ sig, const float* __restrict__ exbuf,
    const float* __restrict__ den, float* __restrict__ node_msg) {
  __shared__ int      s_idx[64];
  __shared__ __align__(16) unsigned s_scal[32 * 64];
  __shared__ float    s_v[32 * 128];
  int lane = threadIdx.x;
  int e0 = blockIdx.x * 32;
  gather_scal32(eidx, e0, lane, Psrc, Ptgt, s_idx, s_scal);

  v16bf aF0[4], aF1[4];
#pragma unroll
  for (int c = 0; c < 4; ++c) {
    aF0[c] = lds_a_frag(s_scal, 0, lane, c);
    aF1[c] = lds_a_frag(s_scal, 16, lane, c);
  }
  int crow = (lane >> 4) * 8, ccol = lane & 15;
  v16bf bp0 = load_b_frag(VcombBf, 64, 0, 0, lane);        // step 0
  v16bf bp1 = load_b_frag(VcombBf, 64, 32, 0, lane);       // step 1
  for (int nt = 0; nt < 8; ++nt) {        // N = 128
    v8f acc0 = {}, acc1 = {};
#pragma unroll
    for (int c = 0; c < 4; ++c) {         // K = 128
      int jn = (nt * 4 + c + 2) & 31;     // step 2 ahead, wrapped
      v16bf bn = load_b_frag(VcombBf, 64, (jn & 3) * 32, (jn >> 2) * 16, lane);
      acc0 = wmma_bf16(aF0[c], bp0, acc0);
      acc1 = wmma_bf16(aF1[c], bp0, acc1);
      bp0 = bp1;
      bp1 = bn;
    }
#pragma unroll
    for (int v = 0; v < 8; ++v) {
      s_v[(crow + v) * 128 + nt * 16 + ccol]      = acc0[v];
      s_v[(16 + crow + v) * 128 + nt * 16 + ccol] = acc1[v];
    }
  }
  __syncthreads();

  float inv_e = 1.0f / N_EDGES;
  float scale = (sig[0] * inv_e) * (sig[1] * inv_e);   // w1 * w2
  for (int i = 0; i < 128; ++i) {         // 32 edges * 128 cols / 32 lanes
    int idx = i * 32 + lane;
    int e = idx >> 7, cc = idx & 127, h = cc >> 4;
    int t = s_idx[32 + e];
    float ex = exbuf[(size_t)(e0 + e) * HEADS + h];
    float al = ex / (den[(size_t)t * HEADS + h] + 1e-16f);
    atomic_add_f32(&node_msg[(size_t)t * SMUL + cc], s_v[e * 128 + cc] * scale * al);
  }
}

// out[:, 0:128] = node_msg @ W_out (WMMA); out[:, 128:480] = 0 (rotation of zeros)
__global__ __launch_bounds__(32) void k_out(const float* __restrict__ node_msg,
                                            const unsigned* __restrict__ WoutBf,
                                            float* __restrict__ out) {
  int lane = threadIdx.x;
  int row0 = blockIdx.x * 16;
  v16bf aF[4];
#pragma unroll
  for (int c = 0; c < 4; ++c) aF[c] = load_a_frag_f32(node_msg, SMUL, row0, c * 32, lane);
  int crow = (lane >> 4) * 8, ccol = lane & 15;
  v16bf bp0 = load_b_frag(WoutBf, 64, 0, 0, lane);
  v16bf bp1 = load_b_frag(WoutBf, 64, 32, 0, lane);
#pragma unroll
  for (int nt = 0; nt < 8; ++nt) {
    v8f acc = {};
#pragma unroll
    for (int c = 0; c < 4; ++c) {
      int jn = (nt * 4 + c + 2) & 31;
      v16bf bn = load_b_frag(WoutBf, 64, (jn & 3) * 32, (jn >> 2) * 16, lane);
      acc = wmma_bf16(aF[c], bp0, acc);
      bp0 = bp1;
      bp1 = bn;
    }
#pragma unroll
    for (int v = 0; v < 8; ++v)
      out[(size_t)(row0 + crow + v) * NODE_DIM + nt * 16 + ccol] = acc[v];
  }
  for (int i = 0; i < 176; ++i) {         // zero 16 rows x 352 cols
    int idx = i * 32 + lane;
    int r = idx / 352, c2 = idx - r * 352;
    out[(size_t)(row0 + r) * NODE_DIM + SMUL + c2] = 0.0f;
  }
}

// ---------------------------------------------------------------------------
// Host launcher
// ---------------------------------------------------------------------------
extern "C" void kernel_launch(void* const* d_in, const int* in_sizes, int n_in,
                              void* d_out, int out_size, void* d_ws, size_t ws_size,
                              hipStream_t stream) {
  (void)in_sizes; (void)n_in; (void)out_size; (void)ws_size;
  const float* nf   = (const float*)d_in[0];
  const int*   eidx = (const int*)  d_in[1];
  const float* ef   = (const float*)d_in[2];
  // d_in[3] batch, d_in[4] wigner_l1, d_in[5] wigner_l2: provably unused
  // (rotations act only on identically-zero l>0 blocks).
  const float* Wmsg = (const float*)d_in[6];
  const float* Wc1  = (const float*)d_in[7];
  const float* ep1w = (const float*)d_in[8];
  const float* ep1b = (const float*)d_in[9];
  const float* Wc2  = (const float*)d_in[10];
  const float* ep2w = (const float*)d_in[11];
  const float* ep2b = (const float*)d_in[12];
  const float* aw   = (const float*)d_in[13];
  const float* ab   = (const float*)d_in[14];
  const float* lng  = (const float*)d_in[15];
  const float* lnb  = (const float*)d_in[16];
  const float* adot = (const float*)d_in[17];
  const float* Wout = (const float*)d_in[18];
  float* out = (float*)d_out;

  char* w = (char*)d_ws;
  auto alloc = [&](size_t bytes) {
    char* p = w;
    w += (bytes + 255) & ~(size_t)255;
    return p;
  };
  float*    Psrc     = (float*)   alloc((size_t)N_NODES * SMUL * 4);
  float*    Ptgt     = (float*)   alloc((size_t)N_NODES * SMUL * 4);
  float*    node_msg = (float*)   alloc((size_t)N_NODES * SMUL * 4);
  float*    logits   = (float*)   alloc((size_t)N_EDGES * HEADS * 4);
  unsigned* nodemax  = (unsigned*)alloc((size_t)N_NODES * HEADS * 4);
  float*    den      = (float*)   alloc((size_t)N_NODES * HEADS * 4);
  unsigned* WmsgBf   = (unsigned*)alloc((size_t)960 * 64 * 4);
  unsigned* WoutBf   = (unsigned*)alloc((size_t)128 * 64 * 4);
  unsigned* AcombBf  = (unsigned*)alloc((size_t)128 * 128 * 4);
  unsigned* VcombBf  = (unsigned*)alloc((size_t)128 * 64 * 4);
  float*    sig      = (float*)   alloc(2 * 4);

  k_init<<<(N_NODES * SMUL + 255) / 256, 256, 0, stream>>>(node_msg, den, nodemax, sig);
  k_pack_bf<<<(960 * 64 + 255) / 256, 256, 0, stream>>>(Wmsg, WmsgBf, 960 * 64);
  k_pack_bf<<<(128 * 64 + 255) / 256, 256, 0, stream>>>(Wout, WoutBf, 128 * 64);
  k_combine<<<(128 * 128 + 128 * 64 + 255) / 256, 256, 0, stream>>>(Wc1, aw, Wc2,
                                                                    AcombBf, VcombBf);
  k_nodeproj<<<N_NODES / 16, 32, 0, stream>>>(nf, WmsgBf, Psrc, Ptgt);
  k_edgescalars<<<(N_EDGES / 32) * 32 / 256, 256, 0, stream>>>(ef, ep1w, ep1b,
                                                               ep2w, ep2b, sig);
  k_edge_logits<<<N_EDGES / 32, 32, 0, stream>>>(eidx, Psrc, Ptgt, AcombBf, ab, lng,
                                                 lnb, adot, sig, logits, nodemax);
  k_den<<<(N_EDGES * HEADS + 255) / 256, 256, 0, stream>>>(eidx, nodemax, logits, den);
  k_edge_accum<<<N_EDGES / 32, 32, 0, stream>>>(eidx, Psrc, Ptgt, VcombBf, sig, logits,
                                                den, node_msg);
  k_out<<<N_NODES / 16, 32, 0, stream>>>(node_msg, WoutBf, out);
}
